// UltraSlumDetectionLoss_70652212019211
// MI455X (gfx1250) — compile-verified, matrix operand order checked by reading, not run on hardware
//
#include <hip/hip_runtime.h>

#define NT 512                        // 16 wave32s per block
#define BATCH 32
#define H 512
#define W 512
#define TILE 64
#define HALO 26                       // 24 chamfer + 1 erode + 1 dilate
#define TDIM (TILE + 2*HALO)          // 116: t buffer
#define EDIM (TDIM - 2)               // 114: edge buffer
#define DDIM (TDIM - 4)               // 112: bmask / distance buffers
#define TILES_X (W / TILE)            // 8
#define TILES_Y (H / TILE)            // 8
#define NTILES (BATCH * TILES_X * TILES_Y)   // 2048
#define NSUMS 6
#define INFD 10000.0f
#define THETA0 3.0f
#define K_CHAMFER 24
#define SQ2f 1.4142135f

typedef float    v2f  __attribute__((ext_vector_type(2)));
typedef float    v8f  __attribute__((ext_vector_type(8)));
typedef _Float16 v16h __attribute__((ext_vector_type(16)));

// ---- wave32 sum reduction via WMMA ----------------------------------------
// A = {v, 0} per lane, B = ones:  D[m][n] = v(lane m) + v(lane m+16).
// Lane L<16 sums D rows 0..7, lane L>=16 rows 8..15; one xor-16 shuffle
// combines the halves so every lane holds the full 32-lane sum.
__device__ __forceinline__ float wave_sum(float v) {
#if __has_builtin(__builtin_amdgcn_wmma_f32_16x16x4_f32)
  v2f a; a.x = v; a.y = 0.0f;
  v2f b; b.x = 1.0f; b.y = 1.0f;
  v8f c = {};
  v8f d = __builtin_amdgcn_wmma_f32_16x16x4_f32(false, a, false, b, (short)0, c,
                                                false, false);
#else
  // fallback: slot0 of 16x32 f16 A maps to (M=lane%16, K=0|8)
  v16h a = {}; a[0] = (_Float16)v;
  v16h b;
#pragma unroll
  for (int i = 0; i < 16; ++i) b[i] = (_Float16)1.0f;
  v8f c = {};
  v8f d = __builtin_amdgcn_wmma_f32_16x16x32_f16(false, a, false, b, (short)0, c,
                                                 false, false);
#endif
  float s = d[0]+d[1]+d[2]+d[3]+d[4]+d[5]+d[6]+d[7];
  s += __shfl_xor(s, 16, 32);
  return s;
}

// ---- gfx1250 async global->LDS copy (ASYNCcnt) ----------------------------
__device__ __forceinline__ void async_ld_b32(unsigned lds_off, const float* gp) {
  unsigned long long ga = (unsigned long long)(const void*)gp;
  asm volatile("global_load_async_to_lds_b32 %0, %1, off"
               :: "v"(lds_off), "v"(ga) : "memory");
}
__device__ __forceinline__ void wait_async0() {
#if __has_builtin(__builtin_amdgcn_s_wait_asynccnt)
  __builtin_amdgcn_s_wait_asynccnt(0);
#else
  asm volatile("s_wait_asynccnt 0" ::: "memory");
#endif
}

__device__ __forceinline__ float softplusf(float z) {
  return (z > 0.0f) ? (z + log1pf(expf(-z))) : log1pf(expf(z));
}

// ---- per-tile kernel: elementwise sums + LDS chamfer boundary term --------
__global__ __launch_bounds__(NT) void ultra_tile_kernel(
    const float* __restrict__ inp, const float* __restrict__ tgt,
    const float* __restrict__ alphap, const float* __restrict__ gammap,
    float* __restrict__ part) {
  __shared__ float sh[TDIM*TDIM + EDIM*EDIM];          // 105,808 B (<320 KB/WG)
  __shared__ float red[(NT/32)*NSUMS];
  float* tb = sh;                // t tile (later: distance ping buffer)
  float* eb = sh + TDIM*TDIM;    // edge   (later: distance pong buffer)

  const int tid = threadIdx.x;
  const int blk = blockIdx.x;
  const int tpi = TILES_X * TILES_Y;
  const int img = blk / tpi;
  const int tix = (blk % tpi) % TILES_X;
  const int tiy = (blk % tpi) / TILES_X;
  const int x0 = tix * TILE - HALO;
  const int y0 = tiy * TILE - HALO;
  const float* __restrict__ timg = tgt + (size_t)img * (H*W);
  const float* __restrict__ ximg = inp + (size_t)img * (H*W);

  // 1) load 116x116 target tile (halo); out-of-image -> 0 (== jnp.pad value)
  const bool inner = (x0 >= 0) && (y0 >= 0) && (x0 + TDIM <= W) && (y0 + TDIM <= H);
  if (inner) {
    const float* base = timg + (size_t)y0 * W + x0;
    for (int k = tid; k < TDIM*TDIM; k += NT) {
      int i = k % TDIM, j = k / TDIM;
      async_ld_b32((unsigned)(unsigned long long)(const void*)&tb[k],
                   base + (size_t)j * W + i);
    }
    wait_async0();
  } else {
    for (int k = tid; k < TDIM*TDIM; k += NT) {
      int i = k % TDIM, j = k / TDIM;
      int gx = x0 + i, gy = y0 + j;
      tb[k] = (gx >= 0 && gx < W && gy >= 0 && gy < H)
                ? timg[(size_t)gy * W + gx] : 0.0f;
    }
  }
  __syncthreads();

  // 2) edge = t * (1 - erode3(t))   (114x114)
  for (int k = tid; k < EDIM*EDIM; k += NT) {
    int i = k % EDIM, j = k / EDIM;
    int c = (j+1)*TDIM + (i+1);
    float m = tb[c];
    m = fminf(m, tb[c-1]);      m = fminf(m, tb[c+1]);
    m = fminf(m, tb[c-TDIM]);   m = fminf(m, tb[c-TDIM-1]); m = fminf(m, tb[c-TDIM+1]);
    m = fminf(m, tb[c+TDIM]);   m = fminf(m, tb[c+TDIM-1]); m = fminf(m, tb[c+TDIM+1]);
    eb[k] = tb[c] * (1.0f - m);
  }
  __syncthreads();

  // 3) d0 = (in-image && dilate3(edge) > 0.5) ? 0 : INF   (112x112, into tb)
  for (int k = tid; k < DDIM*DDIM; k += NT) {
    int i = k % DDIM, j = k / DDIM;
    int c = (j+1)*EDIM + (i+1);
    float m = eb[c];
    m = fmaxf(m, eb[c-1]);      m = fmaxf(m, eb[c+1]);
    m = fmaxf(m, eb[c-EDIM]);   m = fmaxf(m, eb[c-EDIM-1]); m = fmaxf(m, eb[c-EDIM+1]);
    m = fmaxf(m, eb[c+EDIM]);   m = fmaxf(m, eb[c+EDIM-1]); m = fmaxf(m, eb[c+EDIM+1]);
    int gx = x0 + 2 + i, gy = y0 + 2 + j;
    bool inimg = (gx >= 0) && (gx < W) && (gy >= 0) && (gy < H);
    tb[k] = (inimg && m > 0.5f) ? 0.0f : INFD;
  }
  __syncthreads();

  // 4) 24 Jacobi chamfer iterations in LDS (outside buffer == INF)
  float* src = tb; float* dst = eb;
  for (int it = 0; it < K_CHAMFER; ++it) {
    for (int k = tid; k < DDIM*DDIM; k += NT) {
      int i = k % DDIM, j = k / DDIM;
      float dv = src[k];
      if (i > 0)        dv = fminf(dv, src[k-1] + 1.0f);
      if (i < DDIM-1)   dv = fminf(dv, src[k+1] + 1.0f);
      if (j > 0) {
        dv = fminf(dv, src[k-DDIM] + 1.0f);
        if (i > 0)      dv = fminf(dv, src[k-DDIM-1] + SQ2f);
        if (i < DDIM-1) dv = fminf(dv, src[k-DDIM+1] + SQ2f);
      }
      if (j < DDIM-1) {
        dv = fminf(dv, src[k+DDIM] + 1.0f);
        if (i > 0)      dv = fminf(dv, src[k+DDIM-1] + SQ2f);
        if (i < DDIM-1) dv = fminf(dv, src[k+DDIM+1] + SQ2f);
      }
      dst[k] = dv;
    }
    __syncthreads();
    float* t2 = src; src = dst; dst = t2;   // 24 even -> result ends in tb
  }

  // 5) per-pixel terms over the 64x64 interior
  float a = fminf(fmaxf(alphap[0], 0.1f), 0.9f);
  float g = fminf(fmaxf(gammap[0], 1.0f), 5.0f);
  float s1=0, s2=0, s3=0, s4=0, s5=0, s6=0;
  for (int k = tid; k < TILE*TILE; k += NT) {
    int u = k % TILE, vq = k / TILE;
    int gx = tix*TILE + u, gy = tiy*TILE + vq;
    size_t gi = (size_t)gy * W + gx;
    float x = ximg[gi];
    float t = timg[gi];                      // interior re-read: L2-resident
    float p = 1.0f / (1.0f + expf(-x));
    s1 += p * t;  s2 += p;  s3 += t;
    float spn = softplusf(-x);
    float spp = softplusf(x);
    float st  = t * 0.95f + 0.025f;
    float bce_el = st * spn + (1.0f - st) * spp;
    float p_t = st * p + (1.0f - st) * (1.0f - p);
    float a_t = st * a + (1.0f - st) * (1.0f - a);
    s4 += a_t * powf(1.0f - p_t, g) * bce_el;
    s5 += 7.0f * t * spn + (1.0f - t) * spp;
    float dval = src[(vq + (HALO - 2)) * DDIM + (u + (HALO - 2))];
    float wgt  = expf(-fminf(dval, INFD) / THETA0) + 0.1f;
    float pc   = fminf(fmaxf(p, 1e-7f), 1.0f - 1e-7f);
    float bnd  = -(t * logf(pc) + (1.0f - t) * log1pf(-pc));
    s6 += bnd * wgt;
  }

  // 6) block reduction: WMMA wave-reduce + LDS across 16 waves
  float sums[NSUMS] = {s1, s2, s3, s4, s5, s6};
  int wv = tid >> 5, ln = tid & 31;
#pragma unroll
  for (int s = 0; s < NSUMS; ++s) {
    float t2 = wave_sum(sums[s]);            // uniform: EXEC all ones
    if (ln == 0) red[wv * NSUMS + s] = t2;
  }
  __syncthreads();
  if (tid < NSUMS) {
    float tot = 0.0f;
#pragma unroll
    for (int w2 = 0; w2 < NT/32; ++w2) tot += red[w2 * NSUMS + tid];
    part[tid * NTILES + blk] = tot;          // deterministic, no atomics
  }
}

// ---- finalize: reduce 2048 partials per sum, emit scalar loss -------------
__global__ __launch_bounds__(NT) void ultra_final_kernel(
    const float* __restrict__ part, float* __restrict__ out) {
  __shared__ float red[(NT/32)*NSUMS];
  float acc[NSUMS] = {0,0,0,0,0,0};
  for (int b2 = threadIdx.x; b2 < NTILES; b2 += NT) {
#pragma unroll
    for (int s = 0; s < NSUMS; ++s) acc[s] += part[s * NTILES + b2];
  }
  int wv = threadIdx.x >> 5, ln = threadIdx.x & 31;
#pragma unroll
  for (int s = 0; s < NSUMS; ++s) {
    float t2 = wave_sum(acc[s]);
    if (ln == 0) red[wv * NSUMS + s] = t2;
  }
  __syncthreads();
  if (threadIdx.x == 0) {
    float S[NSUMS];
#pragma unroll
    for (int s = 0; s < NSUMS; ++s) {
      float tot = 0.0f;
      for (int w2 = 0; w2 < NT/32; ++w2) tot += red[w2 * NSUMS + s];
      S[s] = tot;
    }
    const float Nf = (float)BATCH * (float)H * (float)W;
    float inter = S[0], sumP = S[1], sumT = S[2];
    float focal    = S[3] / Nf;
    float bce      = S[4] / Nf;
    float boundary = S[5] / Nf;
    float dice_score = (2.0f*inter + 1e-6f) / fmaxf(sumP + sumT + 1e-6f, 1e-7f);
    float dice = (sumT > 0.0f) ? (1.0f - dice_score) : 0.0f;
    float pos_ratio = sumT / Nf;
    float da = 0.7f * (1.0f + pos_ratio);
    float db = 0.3f * (2.0f - pos_ratio);
    float fn = sumT - inter;
    float fp = (sumP - inter) * 3.0f;
    float tversky = 1.0f - (inter + 1e-6f) / (inter + da*fn + db*fp + 1e-6f);
    out[0] = 0.35f*dice + 0.25f*focal + 0.15f*bce + 0.15f*tversky + 0.1f*boundary;
  }
}

extern "C" void kernel_launch(void* const* d_in, const int* in_sizes, int n_in,
                              void* d_out, int out_size, void* d_ws, size_t ws_size,
                              hipStream_t stream) {
  (void)in_sizes; (void)n_in; (void)out_size; (void)ws_size;
  const float* inp    = (const float*)d_in[0];
  const float* tgt    = (const float*)d_in[1];
  const float* alphap = (const float*)d_in[2];
  const float* gammap = (const float*)d_in[3];
  float* part = (float*)d_ws;                 // 6 * 2048 floats = 48 KiB
  float* out  = (float*)d_out;
  ultra_tile_kernel<<<dim3(NTILES), dim3(NT), 0, stream>>>(inp, tgt, alphap, gammap, part);
  ultra_final_kernel<<<dim3(1), dim3(NT), 0, stream>>>(part, out);
}
